// MultiHeadAttention_66056597012603
// MI455X (gfx1250) — matrix operand order repaired and checked by reading, not compiled, not run on hardware
//
#include <hip/hip_runtime.h>
#include <hip/hip_bf16.h>

// MultiHeadAttention forward for MI455X (gfx1250), wave32 + WMMA bf16 + TDM/async-LDS.
// B=2, T=2048, D=1024, H=16, HD=64. Working set ~50MB << 192MB L2 => compute bound
// on matrix throughput: v_wmma_f32_16x16x32_bf16 everywhere, f32 accumulate.

#define BB 2
#define TT 2048
#define DD 1024
#define HH 16
#define HD 64
#define BT (BB * TT)

typedef __bf16 bf16;
typedef __attribute__((ext_vector_type(16))) __bf16 v16bf;
typedef __attribute__((ext_vector_type(8)))  __bf16 v8bf;
typedef __attribute__((ext_vector_type(8)))  float  v8f;
typedef __attribute__((ext_vector_type(4)))  unsigned int v4u;
typedef __attribute__((ext_vector_type(8)))  int v8i;
typedef __attribute__((ext_vector_type(4)))  int v4i;

// ---------------------------------------------------------------------------
// CDNA5 feature detection (compile-safe: degrade to manual staging)
// ---------------------------------------------------------------------------
#if __has_builtin(__builtin_amdgcn_global_load_async_to_lds_b128)
#define HAVE_ASYNC 1
#else
#define HAVE_ASYNC 0
#endif
#if __has_builtin(__builtin_amdgcn_tensor_load_to_lds)
#define HAVE_TDM 1
#else
#define HAVE_TDM 0
#endif

#if HAVE_ASYNC
// Probe-derived prototype: (AS1 v4i* gaddr, AS3 v4i* lds, int offset, int cpol)
typedef __attribute__((address_space(1))) v4i* gp_v4i;
typedef __attribute__((address_space(3))) v4i* lp_v4i;
static __device__ __forceinline__ void async_copy_b128(const void* g, void* l) {
  __builtin_amdgcn_global_load_async_to_lds_b128((gp_v4i)g, (lp_v4i)l, 0, 0);
}
#endif

static __device__ __forceinline__ void wait_async0() {
#if __has_builtin(__builtin_amdgcn_s_wait_asynccnt)
  __builtin_amdgcn_s_wait_asynccnt(0);
#else
  asm volatile("s_wait_asynccnt 0x0" ::: "memory");
#endif
}
static __device__ __forceinline__ void wait_tensor0() {
#if __has_builtin(__builtin_amdgcn_s_wait_tensorcnt)
  __builtin_amdgcn_s_wait_tensorcnt(0);
#else
  asm volatile("s_wait_tensorcnt 0x0" ::: "memory");
#endif
}

#if HAVE_TDM
// Issue TENSOR_LOAD_TO_LDS of a 2D tile [tile_k rows x tile_n cols] of a
// row-major bf16 tensor with row stride `stride` elements, into LDS at `lds`.
// D# packing per CDNA5 ISA 8.3/8.4 (group0: count=1, type=2; group1: data_size=2B).
static __device__ __forceinline__ void tdm_load_2d(const bf16* src, void* lds,
                                                   unsigned tile_n, unsigned tile_k,
                                                   unsigned stride) {
  unsigned long long ga = (unsigned long long)(uintptr_t)src;
  unsigned ldsOff = (unsigned)(uintptr_t)lds;  // generic LDS addr: low 32 bits = offset
  v4u g0;
  g0[0] = 1u;                                   // count=1, user mode
  g0[1] = ldsOff;                               // lds_addr
  g0[2] = (unsigned)ga;                         // global_addr[31:0]
  g0[3] = (unsigned)(ga >> 32) | (2u << 30);    // global_addr[56:32] | type=2
  const unsigned td0 = stride, td1 = 65535u;    // tensor dims (OOB bounds)
  v8i g1;
  g1[0] = (int)(1u << 16);                                  // data_size=2B
  g1[1] = (int)((td0 & 0xffffu) << 16);                     // tensor_dim0[15:0]
  g1[2] = (int)((td0 >> 16) | ((td1 & 0xffffu) << 16));     // dim0 hi / dim1 lo
  g1[3] = (int)((td1 >> 16) | (tile_n << 16));              // dim1 hi / tile_dim0
  g1[4] = (int)tile_k;                                      // tile_dim1 (tile_dim2=0)
  g1[5] = (int)stride;                                      // tensor_dim0_stride lo32
  g1[6] = 0; g1[7] = 0;
  v4i z = {0, 0, 0, 0};
#if __clang_major__ >= 23
  v8i z8 = {0, 0, 0, 0, 0, 0, 0, 0};
  __builtin_amdgcn_tensor_load_to_lds(g0, g1, z, z, z8, 0);
#else
  __builtin_amdgcn_tensor_load_to_lds(g0, g1, z, z, 0);
#endif
}
#endif

// ---------------------------------------------------------------------------
// WMMA helpers (wave32; A and B share contiguous k order k = 16*hi + e)
// ---------------------------------------------------------------------------

static __device__ __forceinline__ v8f wmma_bf16(v16bf a, v16bf b, v8f c) {
  return __builtin_amdgcn_wmma_f32_16x16x32_bf16(false, a, false, b,
                                                 (short)0, c, false, false);
}

// A-fragment (16x32): lane (hi,ln) holds row m0+ln, K runs [kb+hi*16..+15]
// (two contiguous 16B loads).
static __device__ __forceinline__ v16bf load_a_frag(const bf16* __restrict__ base,
                                                    int ld, int m0, int kb) {
  const int lane = threadIdx.x & 31;
  const int hi = lane >> 4;
  const int m  = m0 + (lane & 15);
  const bf16* p = base + (size_t)m * ld + kb + hi * 16;
  v8bf c0 = *(const v8bf*)(p);
  v8bf c1 = *(const v8bf*)(p + 8);
  v16bf a;
#pragma unroll
  for (int i = 0; i < 8; ++i) { a[i] = c0[i]; a[i + 8] = c1[i]; }
  return a;
}

// B-fragment (32x16) from memory laid out mem[n][k] (WMMA-K contiguous).
// Lane (hi,ln): N=n0+ln, K=k0+hi*16+e. Row clamped to [0,nmax).
static __device__ __forceinline__ v16bf load_b_frag(const bf16* __restrict__ rowbase,
                                                    int ld, int n0, int k0, int nmax) {
  const int lane = threadIdx.x & 31;
  const int hi = lane >> 4;
  int n = n0 + (lane & 15);
  n = (n < nmax) ? n : (nmax - 1);
  const bf16* p = rowbase + (size_t)n * ld + k0 + hi * 16;
  v8bf c0 = *(const v8bf*)(p);
  v8bf c1 = *(const v8bf*)(p + 8);
  v16bf b;
#pragma unroll
  for (int i = 0; i < 8; ++i) { b[i] = c0[i]; b[i + 8] = c1[i]; }
  return b;
}

// B-fragment from k-major LDS tile base[k][ldn] (staged untransposed by
// TDM/async engine): strided ds_load_u16 per element.
static __device__ __forceinline__ v16bf load_b_frag_strided(const bf16* base,
                                                            int ldn, int n0, int k0) {
  const int lane = threadIdx.x & 31;
  const int hi = lane >> 4;
  const int n = n0 + (lane & 15);
  v16bf b;
#pragma unroll
  for (int e = 0; e < 16; ++e)
    b[e] = base[(size_t)(k0 + hi * 16 + e) * ldn + n];
  return b;
}

// ---------------------------------------------------------------------------
// Conversion kernels
// ---------------------------------------------------------------------------

__global__ __launch_bounds__(256) void cvt_f32_to_bf16(const float* __restrict__ in,
                                                       bf16* __restrict__ out) {
  const int i = (blockIdx.x * 256 + threadIdx.x) * 4;
  float4 f = *(const float4*)(in + i);
  out[i + 0] = (bf16)f.x;
  out[i + 1] = (bf16)f.y;
  out[i + 2] = (bf16)f.z;
  out[i + 3] = (bf16)f.w;
}

__global__ __launch_bounds__(256) void cvt_transpose_w(const float* __restrict__ w,
                                                       bf16* __restrict__ wt) {
  const int idx = blockIdx.x * 256 + threadIdx.x;
  const int k = idx >> 10;
  const int n = idx & (DD - 1);
  wt[idx] = (bf16)w[(size_t)n * DD + k];
}

// ---------------------------------------------------------------------------
// GEMM: C[BT, D] = A_bf16[BT, D] * Bw_bf16[D, D]
// 256 threads = 8 waves; block tile 16 x 128; wave owns a 16x16 N-tile.
// B chunk (32 x 128) staged in LDS via TDM > async-to-LDS > manual transpose.
// ---------------------------------------------------------------------------

template <int OUT_MODE>  // 0: bf16 -> [B,H,T,HD]; 1: f32 -> [BT,D]
__global__ __launch_bounds__(256) void gemm_bf16_kernel(const bf16* __restrict__ A,
                                                        const bf16* __restrict__ Bw,
                                                        void* __restrict__ out) {
#if HAVE_TDM || HAVE_ASYNC
  __shared__ bf16 ldsB[32][128];   // k-major (as staged by DMA), 8 KB
#else
  __shared__ bf16 ldsBT[128][32];  // n-major (manual transpose), 8 KB
#endif

  const int tid  = threadIdx.x;
  const int wave = tid >> 5;
  const int lane = tid & 31;
  const int hi = lane >> 4;
  const int ln = lane & 15;

  const int m0 = blockIdx.x * 16;
  const int n0 = blockIdx.y * 128;

  v8f acc = {};
  for (int kb = 0; kb < DD; kb += 32) {
    if (kb + 32 < DD)  // global_prefetch_b8
      __builtin_prefetch(Bw + (size_t)(kb + 32) * DD + n0 + (tid & 127), 0, 1);

#if HAVE_TDM
    if (tid < 32) {  // wave 0 drives the Tensor Data Mover
      tdm_load_2d(Bw + (size_t)kb * DD + n0, &ldsB[0][0],
                  /*tile_n=*/128, /*tile_k=*/32, /*stride=*/DD);
      wait_tensor0();
    }
    __syncthreads();
#elif HAVE_ASYNC
    {
      const int kk = tid >> 3;          // 0..31
      const int nn = (tid & 7) * 16;    // 0..112
      const bf16* src = Bw + (size_t)(kb + kk) * DD + n0 + nn;
      bf16* dst = &ldsB[kk][nn];
      async_copy_b128(src, dst);
      async_copy_b128(src + 8, dst + 8);
      wait_async0();
    }
    __syncthreads();
#else
#pragma unroll
    for (int i = 0; i < 16; ++i) {
      const int e = tid + i * 256;
      const int n = e & 127;
      const int k = e >> 7;
      ldsBT[n][k] = Bw[(size_t)(kb + k) * DD + n0 + n];
    }
    __syncthreads();
#endif

    v16bf a = load_a_frag(A, DD, m0, kb);
#if HAVE_TDM || HAVE_ASYNC
    v16bf b = load_b_frag_strided(&ldsB[0][0], 128, wave * 16, 0);
#else
    v16bf b = load_b_frag(&ldsBT[0][0], 32, wave * 16, 0, 128);
#endif
    acc = wmma_bf16(a, b, acc);
    __syncthreads();
  }

  const int n = n0 + wave * 16 + ln;
#pragma unroll
  for (int r = 0; r < 8; ++r) {
    const int m = m0 + r + hi * 8;
    if (OUT_MODE == 0) {
      const int b_ = m / TT, t = m % TT;
      const int h = n / HD, hd = n % HD;
      ((bf16*)out)[(((size_t)b_ * HH + h) * TT + t) * HD + hd] = (bf16)acc[r];
    } else {
      ((float*)out)[(size_t)m * DD + n] = acc[r];
    }
  }
}

// ---------------------------------------------------------------------------
// Flash attention: one wave per 16-query tile of one (b,h).
// Q fragments in registers; K B-fragments contiguous from global; V staged in
// LDS by async-to-LDS (k-major) or manual transpose; P round-trips 1KB LDS.
// ---------------------------------------------------------------------------

__global__ __launch_bounds__(32) void flash_attn_kernel(const bf16* __restrict__ Q,
                                                        const bf16* __restrict__ K,
                                                        const bf16* __restrict__ V,
                                                        bf16* __restrict__ Oa) {
#if HAVE_ASYNC
  __shared__ bf16 ldsV[32][HD];   // [key_local][hd], 4 KB (async staged)
#else
  __shared__ bf16 ldsVT[HD][32];  // [hd][key_local], 4 KB (manual transpose)
#endif
  __shared__ bf16 ldsP[16][32];   // [q_local][key_local], 1 KB

  const int lane = threadIdx.x & 31;
  const int hi = lane >> 4;
  const int ln = lane & 15;

  const int tile = blockIdx.x;
  const int qt = tile % (TT / 16);
  const int bh = tile / (TT / 16);
  const int h  = bh % HH;
  const int b_ = bh / HH;
  const int q0 = qt * 16;

  const bf16* Qb = Q + (size_t)bh * TT * HD;
  const bf16* Kb = K + (size_t)bh * TT * HD;
  const bf16* Vb = V + (size_t)bh * TT * HD;

  v16bf qa0 = load_a_frag(Qb, HD, q0, 0);
  v16bf qa1 = load_a_frag(Qb, HD, q0, 32);

  v8f o0 = {}, o1 = {}, o2 = {}, o3 = {};
  float mrow[8], lrow[8], corrf[8];
#pragma unroll
  for (int r = 0; r < 8; ++r) { mrow[r] = -1.0e30f; lrow[r] = 0.0f; }

  const float scale = 0.125f;   // 1/sqrt(64)
  const int kend = q0 + 16;     // causal

  for (int kb = 0; kb < kend; kb += 32) {
    if (kb + 32 < kend) {
      __builtin_prefetch(Kb + (size_t)(kb + 32 + lane) * HD, 0, 1);
      __builtin_prefetch(Vb + (size_t)(kb + 32 + lane) * HD, 0, 1);
    }

    // ---- stage V block (32 keys x 64 hd) ----
    {
      const int krow = kb + lane;
      const bf16* src = Vb + (size_t)((krow < TT) ? krow : (TT - 1)) * HD;
#if HAVE_ASYNC
      bf16* dst = &ldsV[lane][0];
#pragma unroll
      for (int j = 0; j < 8; ++j)            // 8 x b128 per lane, no VGPR transit
        async_copy_b128(src + j * 8, dst + j * 8);
      wait_async0();
#else
#pragma unroll
      for (int c = 0; c < HD; c += 8) {
        v8bf vv = *(const v8bf*)(src + c);
#pragma unroll
        for (int j = 0; j < 8; ++j) ldsVT[c + j][lane] = vv[j];
      }
#endif
    }

    // ---- scores S = Q * K^T (4 WMMAs) ----
    v8f s0 = {}, s1 = {};
    s0 = wmma_bf16(qa0, load_b_frag(Kb, HD, kb,      0,  TT), s0);
    s0 = wmma_bf16(qa1, load_b_frag(Kb, HD, kb,      32, TT), s0);
    s1 = wmma_bf16(qa0, load_b_frag(Kb, HD, kb + 16, 0,  TT), s1);
    s1 = wmma_bf16(qa1, load_b_frag(Kb, HD, kb + 16, 32, TT), s1);

    // ---- causal mask + online softmax (16-lane butterflies) ----
#pragma unroll
    for (int r = 0; r < 8; ++r) {
      const int qrow = q0 + r + hi * 8;
      const int key0 = kb + ln;
      const int key1 = kb + 16 + ln;
      float x0 = (key0 > qrow) ? -1.0e30f : s0[r] * scale;
      float x1 = (key1 > qrow) ? -1.0e30f : s1[r] * scale;

      float mx = fmaxf(x0, x1);
#pragma unroll
      for (int off = 1; off < 16; off <<= 1) mx = fmaxf(mx, __shfl_xor(mx, off, 32));
      const float mnew = fmaxf(mrow[r], mx);
      const float corr = __expf(mrow[r] - mnew);
      const float p0 = __expf(x0 - mnew);
      const float p1 = __expf(x1 - mnew);
      float ps = p0 + p1;
#pragma unroll
      for (int off = 1; off < 16; off <<= 1) ps += __shfl_xor(ps, off, 32);
      lrow[r] = lrow[r] * corr + ps;
      mrow[r] = mnew;
      corrf[r] = corr;

      ldsP[r + hi * 8][ln]      = (bf16)p0;
      ldsP[r + hi * 8][16 + ln] = (bf16)p1;
    }
#pragma unroll
    for (int r = 0; r < 8; ++r) {
      o0[r] *= corrf[r]; o1[r] *= corrf[r];
      o2[r] *= corrf[r]; o3[r] *= corrf[r];
    }

    // ---- O += P * V (4 WMMAs) ----
    v16bf pa = load_a_frag(&ldsP[0][0], 32, 0, 0);
#if HAVE_ASYNC
    o0 = wmma_bf16(pa, load_b_frag_strided(&ldsV[0][0], HD, 0,  0), o0);
    o1 = wmma_bf16(pa, load_b_frag_strided(&ldsV[0][0], HD, 16, 0), o1);
    o2 = wmma_bf16(pa, load_b_frag_strided(&ldsV[0][0], HD, 32, 0), o2);
    o3 = wmma_bf16(pa, load_b_frag_strided(&ldsV[0][0], HD, 48, 0), o3);
#else
    o0 = wmma_bf16(pa, load_b_frag(&ldsVT[0][0], 32, 0,  0, HD), o0);
    o1 = wmma_bf16(pa, load_b_frag(&ldsVT[0][0], 32, 16, 0, HD), o1);
    o2 = wmma_bf16(pa, load_b_frag(&ldsVT[0][0], 32, 32, 0, HD), o2);
    o3 = wmma_bf16(pa, load_b_frag(&ldsVT[0][0], 32, 48, 0, HD), o3);
#endif
  }

  // ---- epilogue: O /= l, store bf16 into [B,T,D] ----
  bf16* outp = Oa + (size_t)b_ * TT * DD + (size_t)h * HD;
#pragma unroll
  for (int r = 0; r < 8; ++r) {
    const int qrow = q0 + r + hi * 8;
    const float inv = 1.0f / lrow[r];
    bf16* orow = outp + (size_t)qrow * DD;
    orow[ 0 + ln] = (bf16)(o0[r] * inv);
    orow[16 + ln] = (bf16)(o1[r] * inv);
    orow[32 + ln] = (bf16)(o2[r] * inv);
    orow[48 + ln] = (bf16)(o3[r] * inv);
  }
}

// ---------------------------------------------------------------------------
// Launch
// ---------------------------------------------------------------------------

extern "C" void kernel_launch(void* const* d_in, const int* in_sizes, int n_in,
                              void* d_out, int out_size, void* d_ws, size_t ws_size,
                              hipStream_t stream) {
  const float* x  = (const float*)d_in[0];
  const float* wq = (const float*)d_in[1];
  const float* wk = (const float*)d_in[2];
  const float* wv = (const float*)d_in[3];
  const float* wo = (const float*)d_in[4];

  char* ws = (char*)d_ws;
  const size_t MB = 1u << 20;
  bf16* xb   = (bf16*)(ws + 0 * MB);
  bf16* wqT  = (bf16*)(ws + 8 * MB);
  bf16* wkT  = (bf16*)(ws + 10 * MB);
  bf16* wvT  = (bf16*)(ws + 12 * MB);
  bf16* woT  = (bf16*)(ws + 14 * MB);
  bf16* qB   = (bf16*)(ws + 16 * MB);
  bf16* kB   = (bf16*)(ws + 24 * MB);
  bf16* vB   = (bf16*)(ws + 32 * MB);
  bf16* attn = (bf16*)(ws + 40 * MB);

  cvt_f32_to_bf16<<<(BT * DD) / 1024, 256, 0, stream>>>(x, xb);
  cvt_transpose_w<<<(DD * DD) / 256, 256, 0, stream>>>(wq, wqT);
  cvt_transpose_w<<<(DD * DD) / 256, 256, 0, stream>>>(wk, wkT);
  cvt_transpose_w<<<(DD * DD) / 256, 256, 0, stream>>>(wv, wvT);
  cvt_transpose_w<<<(DD * DD) / 256, 256, 0, stream>>>(wo, woT);

  dim3 ggrid(BT / 16, DD / 128);
  gemm_bf16_kernel<0><<<ggrid, 256, 0, stream>>>(xb, wqT, qB);
  gemm_bf16_kernel<0><<<ggrid, 256, 0, stream>>>(xb, wkT, kB);
  gemm_bf16_kernel<0><<<ggrid, 256, 0, stream>>>(xb, wvT, vB);

  flash_attn_kernel<<<BB * HH * (TT / 16), 32, 0, stream>>>(qB, kB, vB, attn);

  gemm_bf16_kernel<1><<<ggrid, 256, 0, stream>>>(attn, woT, (float*)d_out);
}
